// GATv2Model_90675349553671
// MI455X (gfx1250) — matrix operand order repaired and checked by reading, not compiled, not run on hardware
//
#include <hip/hip_runtime.h>
#include <hip/hip_bf16.h>

typedef __attribute__((ext_vector_type(8)))  __bf16 v8bf;
typedef __attribute__((ext_vector_type(16))) __bf16 v16bf;
typedef __attribute__((ext_vector_type(8)))  float  v8f;

#define NNODES 50000
#define NEDGES 800000
#define ETOT   850000   // edges + self loops
#define D1     128      // HEADS*HIDDEN
#define HID    32
#define NGRAPH 64

__device__ __forceinline__ float lrelu(float v) { return v > 0.0f ? v : 0.2f * v; }
__device__ __forceinline__ float elu(float v)   { return v > 0.0f ? v : (__expf(v) - 1.0f); }

// ---------------------------------------------------------------------------
// Workspace init + f32 -> bf16 pre-conversion
// ---------------------------------------------------------------------------
__global__ void fill_f32(float* __restrict__ p, int n, float v) {
    int t = blockIdx.x * blockDim.x + threadIdx.x;
    if (t < n) p[t] = v;
}

__global__ void f32_to_bf16(const float* __restrict__ s, __bf16* __restrict__ d, int n) {
    int i = (blockIdx.x * blockDim.x + threadIdx.x) * 4;   // n is a multiple of 4
    if (i >= n) return;
    float4 v = *(const float4*)(s + i);
    d[i + 0] = (__bf16)v.x;
    d[i + 1] = (__bf16)v.y;
    d[i + 2] = (__bf16)v.z;
    d[i + 3] = (__bf16)v.w;
}

// ---------------------------------------------------------------------------
// Fused WMMA GEMM pair: Y{l,r}[M x N] = Xb[M x 128] @ W{l,r}[N x 128]^T + b{l,r}
// bf16 operands, f32 accumulate, K fully unrolled (4x v_wmma_f32_16x16x32_bf16).
// One wave per 16x16 tile. block=(32, N/16), grid=(M/16, 2) -- blockIdx.y
// selects the left/right transform (shared A operand).
// A layout (16-bit 16x32): lane half -> K offset {0,8}; two 8-consecutive-K
//   groups at +0 and +16  -> two 16B loads.
// B layout (32x16): lane = column N, half -> K offset {0,8}; groups at +0 and +16 -> two 16B loads.
// ---------------------------------------------------------------------------
template <int N>
__global__ void wmma_gemm_pair(const __bf16* __restrict__ Xb,
                               const __bf16* __restrict__ Wl,
                               const __bf16* __restrict__ Wr,
                               const float*  __restrict__ bl,
                               const float*  __restrict__ br,
                               float* __restrict__ Yl,
                               float* __restrict__ Yr) {
    constexpr int K = 128;
    const __bf16* W    = blockIdx.y ? Wr : Wl;
    const float*  bias = blockIdx.y ? br : bl;
    float*        Y    = blockIdx.y ? Yr : Yl;

    const int lane = threadIdx.x & 31;
    const int half = lane >> 4;     // 0: lanes 0-15, 1: lanes 16-31
    const int l16  = lane & 15;
    const int tM   = blockIdx.x;
    const int tN   = threadIdx.y;   // one wave per N tile

    const __bf16* arow = Xb + (size_t)(tM * 16 + l16) * K;   // A row of this lane
    const __bf16* bcol = W  + (size_t)(tN * 16 + l16) * K;   // B column (W[n][k])

    v8f acc = {};
    #pragma unroll
    for (int kb = 0; kb < K; kb += 32) {
        const __bf16* pa = arow + kb + half * 8;
        v8bf a_lo = *(const v8bf*)(pa);
        v8bf a_hi = *(const v8bf*)(pa + 16);
        v16bf a = __builtin_shufflevector(a_lo, a_hi,
                    0, 1, 2, 3, 4, 5, 6, 7, 8, 9, 10, 11, 12, 13, 14, 15);

        const __bf16* pb = bcol + kb + half * 8;
        v8bf b_lo = *(const v8bf*)(pb);
        v8bf b_hi = *(const v8bf*)(pb + 16);
        v16bf b = __builtin_shufflevector(b_lo, b_hi,
                    0, 1, 2, 3, 4, 5, 6, 7, 8, 9, 10, 11, 12, 13, 14, 15);

        acc = __builtin_amdgcn_wmma_f32_16x16x32_bf16(
            /*neg_a=*/false, a, /*neg_b=*/false, b,
            /*c_mod=*/(short)0, acc, /*reuse_a=*/false, /*reuse_b=*/false);
    }

    // D layout: VGPR r -> M = r + 8*half, N = lane&15 (bias is per-lane scalar)
    const float bv = bias[tN * 16 + l16];
    #pragma unroll
    for (int r = 0; r < 8; ++r) {
        int m = tM * 16 + r + 8 * half;
        Y[(size_t)m * N + tN * 16 + l16] = acc[r] + bv;
    }
}

// ---------------------------------------------------------------------------
// Edge helpers: resolve (src,dst) with implicit self loops appended
// ---------------------------------------------------------------------------
__device__ __forceinline__ void edge_pair(const int* __restrict__ ei, int e, int& src, int& dst) {
    if (e < NEDGES) { src = ei[e]; dst = ei[NEDGES + e]; }
    else            { src = dst = e - NEDGES; }
}

// ---- conv1 (4 heads, 32 ch): score + running max -----------------------
__global__ void edge_score1(const int* __restrict__ ei,
                            const float* __restrict__ XL, const float* __restrict__ XR,
                            const float* __restrict__ att,
                            float* __restrict__ score, float* __restrict__ smax) {
    int t = blockIdx.x * blockDim.x + threadIdx.x;
    if (t >= ETOT * 4) return;
    int e = t >> 2, h = t & 3;
    int src, dst; edge_pair(ei, e, src, dst);
    const float* pl = XL + (size_t)src * D1 + h * HID;
    const float* pr = XR + (size_t)dst * D1 + h * HID;
    const float* pa = att + h * HID;
    float s = 0.0f;
    #pragma unroll
    for (int c = 0; c < HID; c += 4) {
        float4 l = *(const float4*)(pl + c);
        float4 r = *(const float4*)(pr + c);
        float4 a = *(const float4*)(pa + c);
        s += a.x * lrelu(l.x + r.x);
        s += a.y * lrelu(l.y + r.y);
        s += a.z * lrelu(l.z + r.z);
        s += a.w * lrelu(l.w + r.w);
    }
    score[(size_t)e * 4 + h] = s;
    atomicMax(smax + (size_t)dst * 4 + h, s);
}

__global__ void edge_exp1(const int* __restrict__ ei,
                          float* __restrict__ score,
                          const float* __restrict__ smax, float* __restrict__ denom) {
    int t = blockIdx.x * blockDim.x + threadIdx.x;
    if (t >= ETOT * 4) return;
    int e = t >> 2, h = t & 3;
    int src, dst; edge_pair(ei, e, src, dst);
    float ex = __expf(score[(size_t)e * 4 + h] - smax[(size_t)dst * 4 + h]);
    score[(size_t)e * 4 + h] = ex;
    atomicAdd(denom + (size_t)dst * 4 + h, ex);
}

__global__ void edge_aggr1(const int* __restrict__ ei,
                           const float* __restrict__ XL,
                           const float* __restrict__ ex, const float* __restrict__ denom,
                           float* __restrict__ out) {
    int t = blockIdx.x * blockDim.x + threadIdx.x;
    if (t >= ETOT * 32) return;           // 32 float4-chunks of 4 ch => 128 ch
    int e = t >> 5, q = t & 31;
    int h = q >> 3, c4 = (q & 7) * 4;
    int src, dst; edge_pair(ei, e, src, dst);
    float alpha = ex[(size_t)e * 4 + h] / (denom[(size_t)dst * 4 + h] + 1e-16f);
    float4 v = *(const float4*)(XL + (size_t)src * D1 + h * HID + c4);
    float* po = out + (size_t)dst * D1 + h * HID + c4;
    atomicAdd(po + 0, v.x * alpha);
    atomicAdd(po + 1, v.y * alpha);
    atomicAdd(po + 2, v.z * alpha);
    atomicAdd(po + 3, v.w * alpha);
}

// h1 = elu(out1 + bias1), in place
__global__ void node_act1(float* __restrict__ out, const float* __restrict__ bias) {
    int t = blockIdx.x * blockDim.x + threadIdx.x;
    if (t >= NNODES * D1) return;
    out[t] = elu(out[t] + bias[t & (D1 - 1)]);
}

// ---- conv2 (1 head, 32 ch) ---------------------------------------------
__global__ void edge_score2(const int* __restrict__ ei,
                            const float* __restrict__ XL, const float* __restrict__ XR,
                            const float* __restrict__ att,
                            float* __restrict__ score, float* __restrict__ smax) {
    int e = blockIdx.x * blockDim.x + threadIdx.x;
    if (e >= ETOT) return;
    int src, dst; edge_pair(ei, e, src, dst);
    const float* pl = XL + (size_t)src * HID;
    const float* pr = XR + (size_t)dst * HID;
    float s = 0.0f;
    #pragma unroll
    for (int c = 0; c < HID; c += 4) {
        float4 l = *(const float4*)(pl + c);
        float4 r = *(const float4*)(pr + c);
        float4 a = *(const float4*)(att + c);
        s += a.x * lrelu(l.x + r.x);
        s += a.y * lrelu(l.y + r.y);
        s += a.z * lrelu(l.z + r.z);
        s += a.w * lrelu(l.w + r.w);
    }
    score[e] = s;
    atomicMax(smax + dst, s);
}

__global__ void edge_exp2(const int* __restrict__ ei,
                          float* __restrict__ score,
                          const float* __restrict__ smax, float* __restrict__ denom) {
    int e = blockIdx.x * blockDim.x + threadIdx.x;
    if (e >= ETOT) return;
    int src, dst; edge_pair(ei, e, src, dst);
    float ex = __expf(score[e] - smax[dst]);
    score[e] = ex;
    atomicAdd(denom + dst, ex);
}

__global__ void edge_aggr2(const int* __restrict__ ei,
                           const float* __restrict__ XL,
                           const float* __restrict__ ex, const float* __restrict__ denom,
                           float* __restrict__ out) {
    int t = blockIdx.x * blockDim.x + threadIdx.x;
    if (t >= ETOT * 8) return;            // 8 float4-chunks => 32 ch
    int e = t >> 3, c4 = (t & 7) * 4;
    int src, dst; edge_pair(ei, e, src, dst);
    float alpha = ex[e] / (denom[dst] + 1e-16f);
    float4 v = *(const float4*)(XL + (size_t)src * HID + c4);
    float* po = out + (size_t)dst * HID + c4;
    atomicAdd(po + 0, v.x * alpha);
    atomicAdd(po + 1, v.y * alpha);
    atomicAdd(po + 2, v.z * alpha);
    atomicAdd(po + 3, v.w * alpha);
}

// h2 = elu(out2 + bias2) and mean-pool accumulation
__global__ void node_act2_pool(const float* __restrict__ out2,
                               const float* __restrict__ bias,
                               const int* __restrict__ batch,
                               float* __restrict__ sums, float* __restrict__ cnt) {
    int t = blockIdx.x * blockDim.x + threadIdx.x;
    if (t >= NNODES * HID) return;
    int i = t >> 5, c = t & 31;
    float v = elu(out2[t] + bias[c]);
    int g = batch[i];
    atomicAdd(sums + (size_t)g * HID + c, v);
    if (c == 0) atomicAdd(cnt + g, 1.0f);
}

// pooled mean + two linear heads; d_out = [class[64] | pseudo[64]]
__global__ void final_heads(const float* __restrict__ sums, const float* __restrict__ cnt,
                            const float* __restrict__ wc, const float* __restrict__ bc,
                            const float* __restrict__ wp, const float* __restrict__ bp,
                            float* __restrict__ out) {
    int g = threadIdx.x;
    if (g >= NGRAPH) return;
    float inv = 1.0f / fmaxf(cnt[g], 1.0f);
    float sc = 0.0f, sp = 0.0f;
    #pragma unroll
    for (int c = 0; c < HID; ++c) {
        float p = sums[(size_t)g * HID + c] * inv;
        sc += p * wc[c];
        sp += p * wp[c];
    }
    out[g]          = sc + bc[0];
    out[NGRAPH + g] = sp + bp[0];
}

// ---------------------------------------------------------------------------
extern "C" void kernel_launch(void* const* d_in, const int* in_sizes, int n_in,
                              void* d_out, int out_size, void* d_ws, size_t ws_size,
                              hipStream_t stream) {
    const float* x     = (const float*)d_in[0];
    const int*   ei    = (const int*)  d_in[1];
    const int*   batch = (const int*)  d_in[2];
    const float* w1l   = (const float*)d_in[3];
    const float* b1l   = (const float*)d_in[4];
    const float* w1r   = (const float*)d_in[5];
    const float* b1r   = (const float*)d_in[6];
    const float* att1  = (const float*)d_in[7];
    const float* bias1 = (const float*)d_in[8];
    const float* w2l   = (const float*)d_in[9];
    const float* b2l   = (const float*)d_in[10];
    const float* w2r   = (const float*)d_in[11];
    const float* b2r   = (const float*)d_in[12];
    const float* att2  = (const float*)d_in[13];
    const float* bias2 = (const float*)d_in[14];
    const float* wc    = (const float*)d_in[15];
    const float* bc    = (const float*)d_in[16];
    const float* wp    = (const float*)d_in[17];
    const float* bp    = (const float*)d_in[18];
    float* out = (float*)d_out;

    // ---- workspace layout ----
    float* ws    = (float*)d_ws;
    float* XL1   = ws;                            // 6,400,000 f32
    float* XR1   = XL1  + (size_t)NNODES * D1;
    float* OUT1  = XR1  + (size_t)NNODES * D1;    // reused as h1
    float* SC1   = OUT1 + (size_t)NNODES * D1;    // 3,400,000
    float* SMAX1 = SC1  + (size_t)ETOT * 4;       // 200,000
    float* DEN1  = SMAX1 + (size_t)NNODES * 4;    // 200,000
    float* XL2   = DEN1  + (size_t)NNODES * 4;    // 1,600,000
    float* XR2   = XL2  + (size_t)NNODES * HID;
    float* OUT2  = XR2  + (size_t)NNODES * HID;   // 1,600,000
    float* SC2   = OUT2 + (size_t)NNODES * HID;   // 850,000
    float* SMAX2 = SC2  + (size_t)ETOT;           // 50,000
    float* DEN2  = SMAX2 + (size_t)NNODES;        // 50,000
    float* SUMS  = DEN2  + (size_t)NNODES;        // 2,048
    float* CNT   = SUMS  + (size_t)NGRAPH * HID;  // 64
    // bf16 region (16B-aligned: float offset so far is a multiple of 4)
    __bf16* Xb   = (__bf16*)(CNT + NGRAPH);       // 6,400,000 bf16
    __bf16* H1b  = Xb   + (size_t)NNODES * D1;    // 6,400,000 bf16
    __bf16* W1LB = H1b  + (size_t)NNODES * D1;    // 16,384
    __bf16* W1RB = W1LB + (size_t)D1 * D1;        // 16,384
    __bf16* W2LB = W1RB + (size_t)D1 * D1;        // 4,096
    __bf16* W2RB = W2LB + (size_t)HID * D1;       // 4,096

    const int TB = 256;
    #define GRID(n) dim3(((n) + TB - 1) / TB)
    #define GRID4(n) dim3((((n) / 4) + TB - 1) / TB)

    // ---- init accumulators ----
    fill_f32<<<GRID(NNODES * D1),  TB, 0, stream>>>(OUT1,  NNODES * D1,  0.0f);
    fill_f32<<<GRID(NNODES * 4),   TB, 0, stream>>>(SMAX1, NNODES * 4,  -3.0e38f);
    fill_f32<<<GRID(NNODES * 4),   TB, 0, stream>>>(DEN1,  NNODES * 4,   0.0f);
    fill_f32<<<GRID(NNODES * HID), TB, 0, stream>>>(OUT2,  NNODES * HID, 0.0f);
    fill_f32<<<GRID(NNODES),       TB, 0, stream>>>(SMAX2, NNODES,      -3.0e38f);
    fill_f32<<<GRID(NNODES),       TB, 0, stream>>>(DEN2,  NNODES,       0.0f);
    fill_f32<<<GRID(NGRAPH * HID), TB, 0, stream>>>(SUMS,  NGRAPH * HID, 0.0f);
    fill_f32<<<GRID(NGRAPH),       TB, 0, stream>>>(CNT,   NGRAPH,       0.0f);

    // ---- pre-convert GEMM operands to bf16 (once) ----
    f32_to_bf16<<<GRID4(NNODES * D1), TB, 0, stream>>>(x,   Xb,   NNODES * D1);
    f32_to_bf16<<<GRID4(D1 * D1),     TB, 0, stream>>>(w1l, W1LB, D1 * D1);
    f32_to_bf16<<<GRID4(D1 * D1),     TB, 0, stream>>>(w1r, W1RB, D1 * D1);
    f32_to_bf16<<<GRID4(HID * D1),    TB, 0, stream>>>(w2l, W2LB, HID * D1);
    f32_to_bf16<<<GRID4(HID * D1),    TB, 0, stream>>>(w2r, W2RB, HID * D1);

    // ---- conv1 dense transforms: fused left/right WMMA GEMM ----
    wmma_gemm_pair<D1><<<dim3(NNODES / 16, 2), dim3(32, D1 / 16), 0, stream>>>(
        Xb, W1LB, W1RB, b1l, b1r, XL1, XR1);

    // ---- conv1 edge phase ----
    edge_score1<<<GRID(ETOT * 4),  TB, 0, stream>>>(ei, XL1, XR1, att1, SC1, SMAX1);
    edge_exp1  <<<GRID(ETOT * 4),  TB, 0, stream>>>(ei, SC1, SMAX1, DEN1);
    edge_aggr1 <<<GRID(ETOT * 32), TB, 0, stream>>>(ei, XL1, SC1, DEN1, OUT1);
    node_act1  <<<GRID(NNODES * D1), TB, 0, stream>>>(OUT1, bias1);

    // ---- conv2 dense transforms ----
    f32_to_bf16<<<GRID4(NNODES * D1), TB, 0, stream>>>(OUT1, H1b, NNODES * D1);
    wmma_gemm_pair<HID><<<dim3(NNODES / 16, 2), dim3(32, HID / 16), 0, stream>>>(
        H1b, W2LB, W2RB, b2l, b2r, XL2, XR2);

    // ---- conv2 edge phase ----
    edge_score2<<<GRID(ETOT),     TB, 0, stream>>>(ei, XL2, XR2, att2, SC2, SMAX2);
    edge_exp2  <<<GRID(ETOT),     TB, 0, stream>>>(ei, SC2, SMAX2, DEN2);
    edge_aggr2 <<<GRID(ETOT * 8), TB, 0, stream>>>(ei, XL2, SC2, DEN2, OUT2);

    // ---- activation + mean pool + heads ----
    node_act2_pool<<<GRID(NNODES * HID), TB, 0, stream>>>(OUT2, bias2, batch, SUMS, CNT);
    final_heads<<<1, NGRAPH, 0, stream>>>(SUMS, CNT, wc, bc, wp, bp, out);

    #undef GRID4
    #undef GRID
}
